// EdgeNet_13108240188000
// MI455X (gfx1250) — matrix operand order repaired and checked by reading, not compiled, not run on hardware
//
#include <hip/hip_runtime.h>
#include <hip/hip_bf16.h>

// Problem constants (match reference)
#define BB 8
#define NN 2000
#define LL 128
#define EE 2048
#define DIN 258           // 2L+2
#define KPAD 288          // DIN padded to multiple of 32 (9 WMMA k-steps)
#define ROWS (BB * NN)    // 16000
#define PENALTY 10.0f

typedef __attribute__((ext_vector_type(16))) __bf16 v16bf;
typedef __attribute__((ext_vector_type(8)))  float  v8f;

struct A8 { uint4 a, b; };  // 32 bytes -> bit_cast to v16bf

__device__ __forceinline__ unsigned short f2bf(float f) {
  unsigned u = __float_as_uint(f);
  unsigned r = u + 0x7FFFu + ((u >> 16) & 1u);   // round-to-nearest-even
  return (unsigned short)(r >> 16);
}

// ---- CDNA5 async global->LDS path (ASYNCcnt), guarded so the file still
// ---- builds as the round-1 version if the builtin is absent.
// ---- Probe result: param type is 'int __vector_size__(16) *' (generic).
#if __has_builtin(__builtin_amdgcn_global_load_async_to_lds_b128)
#define HAVE_ASYNC_LDS 1
typedef int v4i_async __attribute__((vector_size(16)));
__device__ __forceinline__ void async_cp16(const void* g, void* l) {
  // per-lane: LDS[l_lane] <- MEM[g_lane], 16B, tracked by ASYNCcnt
  __builtin_amdgcn_global_load_async_to_lds_b128(
      (v4i_async*)g, (v4i_async*)l, 0, 0);
}
__device__ __forceinline__ void wait_async0() {
#if __has_builtin(__builtin_amdgcn_s_wait_asynccnt)
  __builtin_amdgcn_s_wait_asynccnt(0);
#else
  asm volatile("s_wait_asynccnt 0x0" ::: "memory");
#endif
}
#else
#define HAVE_ASYNC_LDS 0
#endif

// ---------------------------------------------------------------------------
// Kernel 1: fuse W = W1@W2 into packed bf16 B-matrix layout; bias = b1@W2+b2.
// B packed per 32x16 bf16 tile (contiguous k, same as A-frag): lanes 0-15 hold K=0-15 (16 halves),
// lanes 16-31 hold K=16-31; lane = 16*(kk>=16) + (n&15), half index h = kk&15.
// Packed as Bp[((kt*8 + nt)*32 + lane)*16 + h].
// ---------------------------------------------------------------------------
__global__ __launch_bounds__(256) void fuse_kernel(
    const float* __restrict__ W1, const float* __restrict__ b1,
    const float* __restrict__ W2, const float* __restrict__ b2,
    unsigned short* __restrict__ Bp, float* __restrict__ bfused) {
  if (blockIdx.x == 144) {                 // bias block
    int l = threadIdx.x;
    if (l < LL) {
      float s = b2[l];
      for (int e = 0; e < EE; ++e) s += b1[e] * W2[e * LL + l];
      bfused[l] = s;
    }
    return;
  }
  int gid = blockIdx.x * 256 + threadIdx.x;     // 0..36863 = KPAD*LL
  int k = gid >> 7;                             // 0..287
  int l = gid & 127;
  float s = 0.f;
  if (k < DIN) {
    const float* w1r = W1 + (long long)k * EE;
    for (int e = 0; e < EE; ++e) s += w1r[e] * W2[e * LL + l];
  }
  int kt = k >> 5, kk = k & 31, nt = l >> 4;
  int lane = ((kk >> 4) << 4) | (l & 15);
  int h = kk & 15;
  Bp[(((kt * 8 + nt) * 32 + lane) * 16) + h] = f2bf(s);
}

// ---------------------------------------------------------------------------
// Kernel 2: per-row top-128 (smallest, ascending) via u64 bitonic sort in LDS,
// build bf16 A row (KPAD features), save sorted_dist/idx, PENALTY-fill output.
// Dist row staged into LDS with async-to-LDS DMA when available.
// ---------------------------------------------------------------------------
__global__ __launch_bounds__(256) void topk_kernel(
    const float* __restrict__ theta, const float* __restrict__ dist,
    const float* __restrict__ ins0,  const float* __restrict__ ins1,
    unsigned short* __restrict__ X, float* __restrict__ sd,
    int* __restrict__ idxb, float* __restrict__ out) {
  __shared__ unsigned long long keys[2048];
  __shared__ uint4 dstage[NN / 4];                   // 8000 B dist-row stage
  const int tid = threadIdx.x;
  const long long row = blockIdx.x;                  // b*N + n
  const float* drow = dist  + row * NN;
  const float* trow = theta + row * NN;

#if HAVE_ASYNC_LDS
  // DMA the dist row into LDS (ASYNCcnt), then build keys from LDS.
  {
    const uint4* drow4 = (const uint4*)drow;         // 8000B row stride: 16B aligned
    for (int c = tid; c < NN / 4; c += 256) async_cp16(drow4 + c, dstage + c);
  }
#endif
  // Prefetch theta row (gather targets live here) -> global_prefetch_b8
  for (int i = tid; i < 125; i += 256) __builtin_prefetch(trow + i * 16, 0, 0);
  // PENALTY fill of the 2000-wide output row (scatter happens in kernel 3)
  float4 pen = make_float4(PENALTY, PENALTY, PENALTY, PENALTY);
  float4* orow = (float4*)(out + row * NN);
  for (int i = tid; i < NN / 4; i += 256) orow[i] = pen;
#if HAVE_ASYNC_LDS
  wait_async0();
  __syncthreads();
  const float* dsrc = (const float*)dstage;
#else
  const float* dsrc = drow;
#endif

  // Keys: (dist_bits << 32) | idx.  dist > 0 so float bits order == value order.
  for (int i = tid; i < 2048; i += 256) {
    unsigned long long kkey;
    if (i < NN) kkey = ((unsigned long long)__float_as_uint(dsrc[i]) << 32) | (unsigned)i;
    else        kkey = (0xFFFFFFFFull << 32) | (unsigned)i;
    keys[i] = kkey;
  }
  __syncthreads();

  // Bitonic sort, ascending, 2048 keys
  for (unsigned k = 2; k <= 2048; k <<= 1) {
    for (unsigned j = k >> 1; j > 0; j >>= 1) {
      #pragma unroll
      for (int t = 0; t < 8; ++t) {
        unsigned i = (unsigned)(t * 256 + tid);
        unsigned ixj = i ^ j;
        if (ixj > i) {
          unsigned long long a = keys[i], b = keys[ixj];
          bool up = ((i & k) == 0);
          if (up ? (a > b) : (a < b)) { keys[i] = b; keys[ixj] = a; }
        }
      }
      __syncthreads();
    }
  }

  float maxd = __uint_as_float((unsigned)(keys[127] >> 32));
  unsigned short* xrow = X + row * KPAD;
  if (tid < LL) {
    unsigned long long kkey = keys[tid];
    float d = __uint_as_float((unsigned)(kkey >> 32));
    int j = (int)(unsigned)kkey;
    float sdv = d / maxd;
    sd[row * LL + tid]   = sdv;
    idxb[row * LL + tid] = j;
    xrow[tid]      = f2bf(sdv);
    xrow[LL + tid] = f2bf(trow[j]);
  }
  if (tid == 0) { xrow[256] = f2bf(ins0[row]); xrow[257] = f2bf(ins1[row]); }
  if (tid >= 128 && tid < 158) xrow[130 + tid] = 0;   // K pad 258..287 = 0
}

// ---------------------------------------------------------------------------
// Kernel 3: OUT[16000,128] = X[16000,KPAD] @ W + bias - sorted_dist, scattered.
// 128 threads = 4 wave32; each wave owns 16 rows (M=16), N=128 -> 8 N-tiles,
// K loop: 9 x v_wmma_f32_16x16x32_bf16 per N-tile.
// A rows staged with GLOBAL_LOAD_ASYNC_TO_LDS_B128 (ASYNCcnt) when available.
// A-frag, contiguous k order (matches Bp): lane<16 -> m=lane, K 0..15;
// lane>=16 -> m=lane-16, K 16..31.
// ---------------------------------------------------------------------------
__global__ __launch_bounds__(128) void gemm_kernel(
    const unsigned short* __restrict__ X, const unsigned short* __restrict__ Bp,
    const float* __restrict__ bfused, const float* __restrict__ sd,
    const int* __restrict__ idxb, float* __restrict__ out) {
  __shared__ uint4 ldsA4[4 * 576];                   // 4 waves * 16 rows * 576B
  const int tid  = threadIdx.x;
  const int wave = tid >> 5;
  const int lane = tid & 31;
  const long long rowBase = (long long)blockIdx.x * 64 + wave * 16;

  // Stage 16 contiguous A rows (16*576B) into this wave's LDS region
  const uint4* src = (const uint4*)(X + rowBase * KPAD);
  uint4* dst = &ldsA4[wave * 576];
#if HAVE_ASYNC_LDS
  for (int c = lane; c < 576; c += 32) async_cp16(src + c, dst + c);
  wait_async0();
#else
  for (int c = lane; c < 576; c += 32) dst[c] = src[c];
#endif
  __syncthreads();

  const uint* ldsRow = (const uint*)dst;
  const int m   = lane & 15;
  const int hi16 = lane >> 4;                        // 0: lanes 0-15, 1: lanes 16-31
  const int off = hi16 * 8;                          // dword offset into K-tile
  const uint* rowp = ldsRow + m * 144;               // 144 dwords per row (288 bf16)

  // Init accumulators with fused bias (C/D layout: N = lane&15, same for all 8 M rows)
  v8f acc[8];
  #pragma unroll
  for (int nt = 0; nt < 8; ++nt) {
    float bv = bfused[nt * 16 + m];
    #pragma unroll
    for (int r = 0; r < 8; ++r) acc[nt][r] = bv;
  }

  #pragma unroll
  for (int kt = 0; kt < 9; ++kt) {
    A8 a8;
    a8.a = *(const uint4*)(rowp + kt * 16 + off);        // K 0..7   (+16 if hi lanes)
    a8.b = *(const uint4*)(rowp + kt * 16 + 4 + off);    // K 8..15  (+16 if hi lanes)
    v16bf afrag = __builtin_bit_cast(v16bf, a8);
    #pragma unroll
    for (int nt = 0; nt < 8; ++nt) {
      const uint4* bsrc = (const uint4*)(Bp + (((kt * 8 + nt) * 32 + lane) * 16));
      A8 b8; b8.a = bsrc[0]; b8.b = bsrc[1];
      v16bf bfrag = __builtin_bit_cast(v16bf, b8);
      acc[nt] = __builtin_amdgcn_wmma_f32_16x16x32_bf16(
          false, afrag, false, bfrag, (short)0, acc[nt], false, false);
    }
  }

  // Epilogue: D element (vgpr r, lane) -> M = r + 8*hi16, N = nt*16 + (lane&15)
  #pragma unroll
  for (int nt = 0; nt < 8; ++nt) {
    int col = nt * 16 + m;
    #pragma unroll
    for (int r = 0; r < 8; ++r) {
      long long grow = rowBase + r + 8 * hi16;
      float v = acc[nt][r] - sd[grow * LL + col];
      int j = idxb[grow * LL + col];
      out[grow * NN + j] = v;
    }
  }
}

// ---------------------------------------------------------------------------
// Host launch. Inputs (setup_inputs order): theta, dist, ins0, ins1, W1, b1, W2, b2.
// Workspace (~25.7 MB): Bpack | bfused | X(bf16) | sorted_dist | idx
// ---------------------------------------------------------------------------
extern "C" void kernel_launch(void* const* d_in, const int* in_sizes, int n_in,
                              void* d_out, int out_size, void* d_ws, size_t ws_size,
                              hipStream_t stream) {
  (void)in_sizes; (void)n_in; (void)out_size; (void)ws_size;
  const float* theta = (const float*)d_in[0];
  const float* dist  = (const float*)d_in[1];
  const float* ins0  = (const float*)d_in[2];
  const float* ins1  = (const float*)d_in[3];
  const float* W1    = (const float*)d_in[4];
  const float* b1    = (const float*)d_in[5];
  const float* W2    = (const float*)d_in[6];
  const float* b2    = (const float*)d_in[7];
  float* out = (float*)d_out;

  char* ws = (char*)d_ws;
  const size_t OFF_BP  = 0;                       // 9*8*32*16 u16 = 73728 B
  const size_t OFF_BF  = 73728;                   // 128 f32 (rounded)
  const size_t OFF_X   = 74240;                   // 16000*288 u16 = 9,216,000 B
  const size_t OFF_SD  = 9290240;                 // 16000*128 f32 = 8,192,000 B
  const size_t OFF_IDX = 17482240;                // 16000*128 i32 = 8,192,000 B
  unsigned short* Bp  = (unsigned short*)(ws + OFF_BP);
  float*          bf  = (float*)(ws + OFF_BF);
  unsigned short* X   = (unsigned short*)(ws + OFF_X);
  float*          sdw = (float*)(ws + OFF_SD);
  int*            idx = (int*)(ws + OFF_IDX);

  fuse_kernel<<<145, 256, 0, stream>>>(W1, b1, W2, b2, Bp, bf);
  topk_kernel<<<ROWS, 256, 0, stream>>>(theta, dist, ins0, ins1, X, sdw, idx, out);
  gemm_kernel<<<ROWS / 64, 128, 0, stream>>>(X, Bp, bf, sdw, idx, out);
}
